// Encoder_60619168415951
// MI455X (gfx1250) — matrix operand order repaired and checked by reading, not compiled, not run on hardware
//
#include <hip/hip_runtime.h>
#include <hip/hip_bf16.h>
#include <stdint.h>

// ---------------------------------------------------------------------------
// Bidirectional LSTM encoder for MI455X (gfx1250, wave32, WMMA bf16 + TDM).
//   B=32, T=256, E=300 (padded 320), H=512, 4H=2048.
// ---------------------------------------------------------------------------

typedef __attribute__((ext_vector_type(16))) __bf16 v16bf;
typedef __attribute__((ext_vector_type(8)))  float  v8f;
typedef __attribute__((ext_vector_type(4)))  unsigned int u32x4;
typedef __attribute__((ext_vector_type(8)))  int          i32x8;
typedef __attribute__((ext_vector_type(4)))  int          i32x4;

#define B_   32
#define T_   256
#define E_   300
#define EP   320          // E padded to multiple of 32 for K-dim
#define HH   512
#define G4   2048         // 4*H

#if __has_builtin(__builtin_amdgcn_tensor_load_to_lds) && \
    __has_builtin(__builtin_amdgcn_s_wait_tensorcnt)
#define HAVE_TDM 1
#else
#define HAVE_TDM 0
#endif

__device__ __forceinline__ unsigned short f2bf(float f) {
  unsigned u = __float_as_uint(f);
  unsigned r = (u + 0x7FFFu + ((u >> 16) & 1u)) >> 16;   // round-to-nearest-even
  return (unsigned short)r;
}

// ---------------------------------------------------------------------------
// Prep: embedding gather -> bf16 padded Xe[B*T][EP]
// ---------------------------------------------------------------------------
__global__ void k_embed(const int* __restrict__ x, const float* __restrict__ emb,
                        unsigned short* __restrict__ Xe) {
  int i = blockIdx.x * blockDim.x + threadIdx.x;       // over B*T*EP
  if (i >= B_ * T_ * EP) return;
  int row = i / EP, e = i - row * EP;
  float v = 0.f;
  if (e < E_) v = emb[(size_t)x[row] * E_ + e];
  Xe[i] = f2bf(v);
}

// Prep: dst[n][k] = bf16(src[k][n]), k<Ksrc else 0.  src is [Ksrc x 2048].
__global__ void k_transpose_cast(const float* __restrict__ src,
                                 unsigned short* __restrict__ dst,
                                 int Ksrc, int Kpad) {
  int i = blockIdx.x * blockDim.x + threadIdx.x;
  if (i >= G4 * Kpad) return;
  int n = i / Kpad, k = i - n * Kpad;
  float v = (k < Ksrc) ? src[(size_t)k * G4 + n] : 0.f;
  dst[i] = f2bf(v);
}

// Prep: h bf16 double-buffer init (buffer 0 = h0 for each direction)
__global__ void k_hinit(const float* __restrict__ h0f, const float* __restrict__ h0b,
                        unsigned short* __restrict__ hb) {
  int i = blockIdx.x * blockDim.x + threadIdx.x;       // 2 * B * H
  if (i >= 2 * B_ * HH) return;
  int dir = i / (B_ * HH), j = i - dir * (B_ * HH);
  const float* s = dir ? h0b : h0f;
  hb[(size_t)dir * 2 * B_ * HH + j] = f2bf(s[j]);      // buf 0
}

// ---------------------------------------------------------------------------
// Input projection GEMM: xW[dir] = Xe(8192x320,bf16) @ W(bf16) + b, f32 out.
// grid = (2048, 2), 256 thr = 8 waves; each wave -> 16(M) x 64(N) strip.
// ---------------------------------------------------------------------------
__global__ __launch_bounds__(256) void k_ingemm(
    const unsigned short* __restrict__ Xe,
    const unsigned short* __restrict__ WtF, const unsigned short* __restrict__ WtB,
    const float* __restrict__ bF, const float* __restrict__ bB,
    float* __restrict__ xWF, float* __restrict__ xWB) {
  const int dir = blockIdx.y;
  const unsigned short* Wt = dir ? WtB : WtF;
  const float* bias = dir ? bB : bF;
  float* xW = dir ? xWB : xWF;

  const int wid  = threadIdx.x >> 5;
  const int lane = threadIdx.x & 31;
  const int hs = lane >> 4, ln = lane & 15;
  const int w  = blockIdx.x * 8 + wid;   // strip id, 0..16383
  const int m0 = (w >> 5) * 16;
  const int n0 = (w & 31) * 64;

  v8f acc[4];
#pragma unroll
  for (int i = 0; i < 4; ++i) {
    float bv = bias[n0 + i * 16 + ln];   // bias depends only on N = lane&15
#pragma unroll
    for (int r = 0; r < 8; ++r) acc[i][r] = bv;
  }

  const unsigned short* arow = Xe + (size_t)(m0 + ln) * EP;
  for (int kk = 0; kk < EP; kk += 32) {
    union { uint4 u[2]; v16bf v; } A;
    const unsigned short* ap = arow + kk + hs * 16;
    A.u[0] = *(const uint4*)(ap);
    A.u[1] = *(const uint4*)(ap + 8);
    union { uint4 u[2]; v16bf v; } Bm[4];
#pragma unroll
    for (int i = 0; i < 4; ++i) {
      const unsigned short* bp = Wt + (size_t)(n0 + i * 16 + ln) * EP + kk + hs * 16;
      Bm[i].u[0] = *(const uint4*)(bp);
      Bm[i].u[1] = *(const uint4*)(bp + 8);
    }
#pragma unroll
    for (int i = 0; i < 4; ++i)
      acc[i] = __builtin_amdgcn_wmma_f32_16x16x32_bf16(false, A.v, false, Bm[i].v,
                                                       (short)0, acc[i], false, false);
  }
#pragma unroll
  for (int i = 0; i < 4; ++i)
#pragma unroll
    for (int r = 0; r < 8; ++r)
      xW[(size_t)(m0 + hs * 8 + r) * G4 + (n0 + i * 16 + ln)] = acc[i][r];
}

// ---------------------------------------------------------------------------
// Persistent recurrent kernel. grid = 16 blocks: blockIdx.x = dir*8 + wg.
// Block owns hidden cols [wg*64, wg*64+64) -> 256 gate cols. One global
// barrier per step among the 8 blocks of a direction; h is double-buffered.
// TDM streams each step's xW gate chunk (32 KB) into LDS under the GEMM.
// ---------------------------------------------------------------------------
__global__ __launch_bounds__(256) void k_lstm(
    const int* __restrict__ x,
    const float* __restrict__ xWF, const float* __restrict__ xWB,
    const unsigned short* __restrict__ UtF, const unsigned short* __restrict__ UtB,
    const float* __restrict__ h0F, const float* __restrict__ c0F,
    const float* __restrict__ h0B, const float* __restrict__ c0B,
    unsigned short* __restrict__ hb,        // [2 dirs][2 bufs][B*H] bf16
    unsigned int* __restrict__ bar,         // [2 dirs][T] counters (zeroed)
    float* __restrict__ out) {
  __shared__ float zl[B_][256];             // 32 KB: z chunk (4 gates x 64)
  __shared__ float cl[B_][64];              // 8 KB: f32 cell-state chunk
  __shared__ float hl[B_][64];              // 8 KB: f32 hidden chunk (carry)
  __shared__ float xwl[B_][256];            // 32 KB: TDM-staged xW gate chunk

  const int dir = blockIdx.x >> 3;
  const int wg  = blockIdx.x & 7;
  const float* xW = dir ? xWB : xWF;
  const unsigned short* Ut = dir ? UtB : UtF;
  const float* h0 = dir ? h0B : h0F;
  const float* c0 = dir ? c0B : c0F;
  unsigned short* hbuf = hb + (size_t)dir * 2 * B_ * HH;
  unsigned int* mybar = bar + dir * T_;

  for (int i = threadIdx.x; i < B_ * 64; i += 256) {
    int b = i >> 6, hc = i & 63;
    hl[b][hc] = h0[b * HH + wg * 64 + hc];
    cl[b][hc] = c0[b * HH + wg * 64 + hc];
  }
  __syncthreads();

  const int wid  = threadIdx.x >> 5;
  const int lane = threadIdx.x & 31;
  const int hs = lane >> 4, ln = lane & 15;
  const int mtl = (wid & 1) * 16;           // M-tile base (B=32 -> 2 tiles)
  const int n0l = (wid >> 1) * 64;          // local N strip (of 256)

#if HAVE_TDM
  // Step-invariant pieces of the Tensor DMA descriptor (ISA Ch.8).
  const unsigned lds_off = (unsigned)(size_t)&xwl[0][0]; // low 32b = LDS addr
  i32x8 g1;
  g1[0] = (int)(2u << 16);                         // wg_mask=0, data_size=4B
  g1[1] = 0;                                       // no atomic barrier; td0 lo16=0
  g1[2] = 0x4000;                                  // tensor_dim0=2^30 (hi16)
  g1[3] = (int)(0x4000u | (64u << 16));            // tensor_dim1=2^30; tile_dim0=64
  g1[4] = (int)(4u | (32u << 16));                 // tile_dim1=4 (gates); tile_dim2=32 (b)
  g1[5] = HH;                                      // dim0_stride = 512 (gate step)
  g1[6] = 0;                                       // dim0_stride hi; dim1_stride lo16
  g1[7] = (int)(((unsigned)(T_ * G4)) >> 16);      // dim1_stride = T*2048 (batch step)
  i32x4 g2; g2[0] = 0x40000000; g2[1] = 0; g2[2] = 0; g2[3] = 0;  // tensor_dim2 big
  i32x4 g3; g3[0] = 0; g3[1] = 0; g3[2] = 0; g3[3] = 0;
  i32x8 g4; g4[0] = 0; g4[1] = 0; g4[2] = 0; g4[3] = 0;           // extended-form
  g4[4] = 0; g4[5] = 0; g4[6] = 0; g4[7] = 0;                     // words (unused)
#endif

  union Frag { uint4 u[2]; v16bf v; };

  for (int t = 0; t < T_; ++t) {
    const int tt = dir ? (T_ - 1 - t) : t;
    const unsigned short* hcur = hbuf + (size_t)(t & 1) * B_ * HH;

#if HAVE_TDM
    {   // kick off async DMA of this step's xW chunk -> xwl (overlaps GEMM)
      unsigned long long ga =
          (unsigned long long)(const void*)(xW + (size_t)tt * G4 + wg * 64);
      u32x4 g0;
      g0[0] = 1u;                                  // count=1 (valid descriptor)
      g0[1] = lds_off;
      g0[2] = (unsigned)(ga & 0xFFFFFFFFu);
      g0[3] = (unsigned)((ga >> 32) & 0x1FFFFFFu) | (2u << 30);  // type=2
      if (threadIdx.x == 0)
        __builtin_amdgcn_tensor_load_to_lds(g0, g1, g2, g3, g4, 0);
    }
#endif

    // ---- z_chunk = h @ U[:, block cols]  (K = 512, 16 WMMA steps/tile) ----
    v8f acc[4];
#pragma unroll
    for (int i = 0; i < 4; ++i)
#pragma unroll
      for (int r = 0; r < 8; ++r) acc[i][r] = 0.f;

    const unsigned short* arow = hcur + (size_t)(mtl + ln) * HH;
    Frag A[2], Bm[2][4];
    auto loadA = [&](Frag& d, int kk) {
      const unsigned short* ap = arow + kk + hs * 16;
      d.u[0] = *(const uint4*)(ap);
      d.u[1] = *(const uint4*)(ap + 8);
    };
    auto loadB = [&](Frag* d, int kk) {
#pragma unroll
      for (int i = 0; i < 4; ++i) {
        int nloc = n0l + i * 16 + ln;                        // 0..255
        int ncol = (nloc >> 6) * HH + wg * 64 + (nloc & 63); // gate*512+hc
        const unsigned short* bp = Ut + (size_t)ncol * HH + kk + hs * 16;
        d[i].u[0] = *(const uint4*)(bp);
        d[i].u[1] = *(const uint4*)(bp + 8);
      }
    };
    loadA(A[0], 0);
    loadB(Bm[0], 0);
#pragma unroll
    for (int kk = 0; kk < 16; ++kk) {      // software-pipelined: prefetch k+1
      const int cur = kk & 1, nxt = cur ^ 1;
      if (kk < 15) {
        loadA(A[nxt], (kk + 1) * 32);
        loadB(Bm[nxt], (kk + 1) * 32);
      }
#pragma unroll
      for (int i = 0; i < 4; ++i)
        acc[i] = __builtin_amdgcn_wmma_f32_16x16x32_bf16(false, A[cur].v, false,
                                                         Bm[cur][i].v, (short)0,
                                                         acc[i], false, false);
    }
#pragma unroll
    for (int i = 0; i < 4; ++i)
#pragma unroll
      for (int r = 0; r < 8; ++r)
        zl[mtl + hs * 8 + r][n0l + i * 16 + ln] = acc[i][r];

#if HAVE_TDM
    if (wid == 0) __builtin_amdgcn_s_wait_tensorcnt(0);  // drain DMA (wave 0)
#endif
    __syncthreads();

    // ---- gates + state update (f32), masked carry, write h/out ----
    unsigned short* hnext = hbuf + (size_t)((t + 1) & 1) * B_ * HH;
    const bool last = (t == T_ - 1);
    for (int i = threadIdx.x; i < B_ * 64; i += 256) {
      int b = i >> 6, hcl = i & 63;
      int hc = wg * 64 + hcl;
#if HAVE_TDM
      float zi = xwl[b][hcl]       + zl[b][hcl];
      float zf = xwl[b][64 + hcl]  + zl[b][64 + hcl];
      float zg = xwl[b][128 + hcl] + zl[b][128 + hcl];
      float zo = xwl[b][192 + hcl] + zl[b][192 + hcl];
#else
      const float* xwrow = xW + (size_t)(b * T_ + tt) * G4;
      float zi = xwrow[0 * HH + hc] + zl[b][hcl];
      float zf = xwrow[1 * HH + hc] + zl[b][64 + hcl];
      float zg = xwrow[2 * HH + hc] + zl[b][128 + hcl];
      float zo = xwrow[3 * HH + hc] + zl[b][192 + hcl];
#endif
      float ig = 1.f / (1.f + expf(-zi));
      float fg = 1.f / (1.f + expf(-zf));
      float gg = tanhf(zg);
      float og = 1.f / (1.f + expf(-zo));
      float cp = cl[b][hcl], hp = hl[b][hcl];
      float cn = fg * cp + ig * gg;
      float hn = og * tanhf(cn);
      bool m = (x[b * T_ + tt] != 0);
      float hv = m ? hn : hp;
      float cv = m ? cn : cp;
      cl[b][hcl] = cv;
      hl[b][hcl] = hv;
      out[(size_t)(b * T_ + tt) * 1024 + dir * HH + hc] = hv;
      hnext[b * HH + hc] = f2bf(hv);
      if (last) {
        out[8388608 + dir * 16384 + b * HH + hc] = hv;          // h_f / h_b
        out[8388608 + 32768 + dir * 16384 + b * HH + hc] = cv;  // c_f / c_b
      }
    }
    __syncthreads();

    // ---- inter-block barrier (8 blocks of this direction) ----
    if (threadIdx.x == 0) {
      __threadfence();
      __hip_atomic_fetch_add(&mybar[t], 1u, __ATOMIC_RELEASE, __HIP_MEMORY_SCOPE_AGENT);
      while (__hip_atomic_load(&mybar[t], __ATOMIC_ACQUIRE, __HIP_MEMORY_SCOPE_AGENT) < 8u)
        __builtin_amdgcn_s_sleep(1);
    }
    __syncthreads();
    __threadfence();   // acquire: make other blocks' h writes visible
  }
}

// ---------------------------------------------------------------------------
// Host launcher
// ---------------------------------------------------------------------------
extern "C" void kernel_launch(void* const* d_in, const int* in_sizes, int n_in,
                              void* d_out, int out_size, void* d_ws, size_t ws_size,
                              hipStream_t stream) {
  (void)in_sizes; (void)n_in; (void)out_size; (void)ws_size;
  const int*   x   = (const int*)  d_in[0];
  const float* h0f = (const float*)d_in[1];
  const float* c0f = (const float*)d_in[2];
  const float* h0b = (const float*)d_in[3];
  const float* c0b = (const float*)d_in[4];
  const float* emb = (const float*)d_in[5];
  const float* Wf  = (const float*)d_in[6];
  const float* Uf  = (const float*)d_in[7];
  const float* bf  = (const float*)d_in[8];
  const float* Wb  = (const float*)d_in[9];
  const float* Ub  = (const float*)d_in[10];
  const float* bb  = (const float*)d_in[11];
  float* out = (float*)d_out;

  // workspace layout
  char* ws = (char*)d_ws;
  const size_t SZ_BAR = 4096;
  const size_t SZ_XE  = (size_t)B_ * T_ * EP * 2;     // 5,242,880
  const size_t SZ_WT  = (size_t)G4 * EP * 2;          // 1,310,720
  const size_t SZ_UT  = (size_t)G4 * HH * 2;          // 2,097,152
  const size_t SZ_HB  = (size_t)2 * 2 * B_ * HH * 2;  // 131,072
  const size_t SZ_XW  = (size_t)B_ * T_ * G4 * 4;     // 67,108,864

  unsigned int*   bar = (unsigned int*)ws;
  unsigned short* Xe  = (unsigned short*)(ws + SZ_BAR);
  unsigned short* WtF = (unsigned short*)((char*)Xe  + SZ_XE);
  unsigned short* WtB = (unsigned short*)((char*)WtF + SZ_WT);
  unsigned short* UtF = (unsigned short*)((char*)WtB + SZ_WT);
  unsigned short* UtB = (unsigned short*)((char*)UtF + SZ_UT);
  unsigned short* hbf = (unsigned short*)((char*)UtB + SZ_UT);
  float*          xWF = (float*)((char*)hbf + SZ_HB);
  float*          xWB = (float*)((char*)xWF + SZ_XW);

  // barrier counters must start at 0 every call (ws is not re-poisoned)
  (void)hipMemsetAsync(bar, 0, SZ_BAR, stream);

  k_embed<<<(B_ * T_ * EP + 255) / 256, 256, 0, stream>>>(x, emb, Xe);
  k_transpose_cast<<<(G4 * EP + 255) / 256, 256, 0, stream>>>(Wf, WtF, E_, EP);
  k_transpose_cast<<<(G4 * EP + 255) / 256, 256, 0, stream>>>(Wb, WtB, E_, EP);
  k_transpose_cast<<<(G4 * HH + 255) / 256, 256, 0, stream>>>(Uf, UtF, HH, HH);
  k_transpose_cast<<<(G4 * HH + 255) / 256, 256, 0, stream>>>(Ub, UtB, HH, HH);
  k_hinit<<<(2 * B_ * HH + 255) / 256, 256, 0, stream>>>(h0f, h0b, hbf);

  dim3 g2(2048, 2);
  k_ingemm<<<g2, 256, 0, stream>>>(Xe, WtF, WtB, bf, bb, xWF, xWB);

  k_lstm<<<16, 256, 0, stream>>>(x, xWF, xWB, UtF, UtB,
                                 h0f, c0f, h0b, c0b, hbf, bar, out);
}
